// GCN_82746839924853
// MI455X (gfx1250) — matrix operand order repaired and checked by reading, not compiled, not run on hardware
//
#include <hip/hip_runtime.h>
#include <hip/hip_bf16.h>

#define NN      100000
#define NE      1600000
#define NG      512
#define DIN     192
#define DH      64
#define BN_EPS  1e-5f

typedef __attribute__((ext_vector_type(16))) _Float16 v16h;
typedef __attribute__((ext_vector_type(8)))  _Float16 v8h;
typedef __attribute__((ext_vector_type(8)))  float    v8f;

static __device__ __forceinline__ float sigmoidf_(float x) {
    return 1.0f / (1.0f + __expf(-x));
}

// ordered-uint encoding so atomicMax(unsigned) == float max
static __device__ __forceinline__ unsigned encf(float f) {
    unsigned u = __float_as_uint(f);
    return ((int)u >= 0) ? (u | 0x80000000u) : ~u;
}
static __device__ __forceinline__ float decf(unsigned u) {
    return (u & 0x80000000u) ? __uint_as_float(u & 0x7FFFFFFFu) : __uint_as_float(~u);
}

// ---------------- utility kernels ----------------

__global__ void zero_f32(float* __restrict__ p, int n) {
    int i = blockIdx.x * blockDim.x + threadIdx.x;
    if (i < n) p[i] = 0.0f;
}

__global__ void deg_init(float* __restrict__ deg) {
    int i = blockIdx.x * blockDim.x + threadIdx.x;
    if (i < NN) deg[i] = 1.0f;   // self loop
}

__global__ void deg_count(const int* __restrict__ dst, float* __restrict__ deg) {
    int e = blockIdx.x * blockDim.x + threadIdx.x;
    if (e < NE) atomicAdd(&deg[dst[e]], 1.0f);
}

__global__ void deg_rsqrt(float* __restrict__ deg) {
    int i = blockIdx.x * blockDim.x + threadIdx.x;
    if (i < NN) deg[i] = rsqrtf(deg[i]);
}

// Embedding gather: x[N,6] -> concat f16 features [N,192]
__global__ void embed_kernel(const int* __restrict__ x,
                             const float* __restrict__ e0, const float* __restrict__ e1,
                             const float* __restrict__ e2, const float* __restrict__ e3,
                             const float* __restrict__ e4, const float* __restrict__ e5,
                             _Float16* __restrict__ xf) {
    long idx = (long)blockIdx.x * blockDim.x + threadIdx.x;
    if (idx >= (long)NN * DIN) return;
    int i = (int)(idx / DIN);
    int j = (int)(idx % DIN);
    int t = j >> 5, col = j & 31;
    const float* tabs[6] = { e0, e1, e2, e3, e4, e5 };
    int row = x[i * 6 + t];
    xf[idx] = (_Float16)tabs[t][row * 32 + col];
}

// Pack W[K,64] (f32 row-major) into WMMA B-fragment layout:
// fragment f = (ntile*(K/32)+kb), lane, element j -> B[kb*32 + (j>>3)*16 + (lane>>4)*8 + (j&7)][ntile*16 + lane&15]
__global__ void pack_b(const float* __restrict__ W, _Float16* __restrict__ Bp, int K) {
    int idx = blockIdx.x * blockDim.x + threadIdx.x;
    int total = 64 * K;            // 4 * (K/32) * 32 * 16
    if (idx >= total) return;
    int j    = idx & 15;
    int lane = (idx >> 4) & 31;
    int rest = idx >> 9;
    int kblocks = K >> 5;
    int kb    = rest % kblocks;
    int ntile = rest / kblocks;
    int n  = lane & 15;
    int kk = kb * 32 + ((j >> 3) << 4) + ((lane >> 4) << 3) + (j & 7);
    Bp[idx] = (_Float16)W[(size_t)kk * DH + (ntile << 4) + n];
}

// ---------------- WMMA GEMM: out[N,64] = A[N,K](f16) @ W[K,64] (+bias) ----------------
template <int K>
__global__ __launch_bounds__(128)
void wmma_gemm_kernel(const _Float16* __restrict__ A,
                      const _Float16* __restrict__ Bp,
                      const float* __restrict__ bias,
                      float* __restrict__ out) {
    const int lane  = threadIdx.x & 31;
    const int ntile = threadIdx.x >> 5;      // 0..3 -> output cols [16*ntile, 16*ntile+16)
    const int m0    = blockIdx.x << 4;       // 16-row tile
    const int mrow  = m0 + (lane & 15);
    const int khalf = lane >> 4;             // 0 or 1

    v8f c = { 0.f, 0.f, 0.f, 0.f, 0.f, 0.f, 0.f, 0.f };
    const _Float16* arow = A + (size_t)mrow * K + khalf * 8;
    const v16h* bptr = (const v16h*)Bp + (size_t)(ntile * (K / 32)) * 32 + lane;

#pragma unroll
    for (int kb = 0; kb < K / 32; ++kb) {
        union { v16h v; v8h h[2]; } a;
        a.h[0] = *(const v8h*)(arow + kb * 32);        // K = kb*32 + khalf*8 + [0..7]
        a.h[1] = *(const v8h*)(arow + kb * 32 + 16);   // K = kb*32 + 16 + khalf*8 + [0..7]
        v16h b = bptr[kb * 32];
        c = __builtin_amdgcn_wmma_f32_16x16x32_f16(false, a.v, false, b,
                                                   (short)0, c, false, false);
    }

    const int ncol = (ntile << 4) + (lane & 15);
    const int mb   = m0 + (khalf << 3);
    float bb = bias ? bias[ncol] : 0.0f;
#pragma unroll
    for (int r = 0; r < 8; ++r)
        out[(size_t)(mb + r) * DH + ncol] = c[r] + bb;
}

// ---------------- GCN scatter ----------------

// agg[i,c] = dinv[i]^2 * hw[i,c]   (self-loop term; also initializes agg)
__global__ void selfloop_kernel(const float* __restrict__ hw, const float* __restrict__ dinv,
                                float* __restrict__ agg) {
    long idx = (long)blockIdx.x * blockDim.x + threadIdx.x;
    if (idx >= (long)NN * DH) return;
    int i = (int)(idx >> 6);
    float di = dinv[i];
    agg[idx] = di * di * hw[idx];
}

// agg[dst,c] += dinv[src]*dinv[dst]*hw[src,c]
__global__ void edge_scatter(const int* __restrict__ src, const int* __restrict__ dst,
                             const float* __restrict__ hw, const float* __restrict__ dinv,
                             float* __restrict__ agg) {
    long gid = (long)blockIdx.x * blockDim.x + threadIdx.x;
    if (gid >= (long)NE * DH) return;
    int e = (int)(gid >> 6);
    int c = (int)(gid & 63);
    int s = src[e], d = dst[e];
    float nrm = dinv[s] * dinv[d];
    atomicAdd(&agg[(size_t)d * DH + c], nrm * hw[(size_t)s * DH + c]);
}

// h = sigmoid(agg + bias); write f32 and f16 copies
__global__ void bias_sigmoid(const float* __restrict__ agg, const float* __restrict__ bias,
                             float* __restrict__ h32, _Float16* __restrict__ h16) {
    long idx = (long)blockIdx.x * blockDim.x + threadIdx.x;
    if (idx >= (long)NN * DH) return;
    int c = (int)(idx & 63);
    float v = sigmoidf_(agg[idx] + bias[c]);
    h32[idx] = v;
    h16[idx] = (_Float16)v;
}

// ---------------- BatchNorm (training stats) ----------------

__global__ void bn_partial(const float* __restrict__ g, float* __restrict__ sum,
                           float* __restrict__ sumsq) {
    int t = blockIdx.x * blockDim.x + threadIdx.x;
    int c = t & 63;
    int r0 = t >> 6;
    int rs = (gridDim.x * blockDim.x) >> 6;
    float s = 0.f, ss = 0.f;
    for (int i = r0; i < NN; i += rs) {
        float v = g[(size_t)i * DH + c];
        s += v; ss += v * v;
    }
    atomicAdd(&sum[c], s);
    atomicAdd(&sumsq[c], ss);
}

__global__ void bn_finalize(const float* __restrict__ sum, const float* __restrict__ sumsq,
                            const float* __restrict__ gamma, const float* __restrict__ beta,
                            float* __restrict__ scale, float* __restrict__ shift) {
    int c = threadIdx.x;
    if (c >= DH) return;
    float mu  = sum[c] / (float)NN;
    float var = sumsq[c] / (float)NN - mu * mu;
    float sc  = gamma[c] * rsqrtf(var + BN_EPS);
    scale[c] = sc;
    shift[c] = beta[c] - mu * sc;
}

// gate[i] = relu(BN(g[i,:])) . gate_w2 + gate_b2
__global__ void gate_kernel(const float* __restrict__ g, const float* __restrict__ scale,
                            const float* __restrict__ shift, const float* __restrict__ w2,
                            const float* __restrict__ b2, float* __restrict__ gate) {
    int i = blockIdx.x * blockDim.x + threadIdx.x;
    if (i >= NN) return;
    float acc = 0.f;
#pragma unroll
    for (int c = 0; c < DH; ++c) {
        float v = g[(size_t)i * DH + c] * scale[c] + shift[c];
        v = fmaxf(v, 0.f);
        acc += v * w2[c];
    }
    gate[i] = acc + b2[0];
}

// ---------------- segment softmax + pooling ----------------

__global__ void gmax_kernel(const float* __restrict__ gate, const int* __restrict__ batch,
                            unsigned* __restrict__ gmaxU) {
    int i = blockIdx.x * blockDim.x + threadIdx.x;
    if (i >= NN) return;
    atomicMax(&gmaxU[batch[i]], encf(gate[i]));
}

__global__ void expz_kernel(const float* __restrict__ gate, const int* __restrict__ batch,
                            const unsigned* __restrict__ gmaxU, float* __restrict__ ebuf,
                            float* __restrict__ z) {
    int i = blockIdx.x * blockDim.x + threadIdx.x;
    if (i >= NN) return;
    int b = batch[i];
    float e = __expf(gate[i] - decf(gmaxU[b]));
    ebuf[i] = e;
    atomicAdd(&z[b], e);
}

__global__ void pooled_kernel(const float* __restrict__ ebuf, const float* __restrict__ z,
                              const int* __restrict__ batch, const float* __restrict__ h,
                              float* __restrict__ pooled) {
    long idx = (long)blockIdx.x * blockDim.x + threadIdx.x;
    if (idx >= (long)NN * DH) return;
    int i = (int)(idx >> 6);
    int c = (int)(idx & 63);
    int b = batch[i];
    float alpha = ebuf[i] / z[b];
    atomicAdd(&pooled[(size_t)b * DH + c], alpha * h[idx]);
}

__global__ void final_kernel(const float* __restrict__ pooled, const float* __restrict__ lw,
                             const float* __restrict__ lb, float* __restrict__ out) {
    int g = blockIdx.x * blockDim.x + threadIdx.x;
    if (g >= NG) return;
    float acc = lb[0];
#pragma unroll
    for (int c = 0; c < DH; ++c) acc += pooled[(size_t)g * DH + c] * lw[c];
    out[g] = sigmoidf_(acc);
}

// ---------------- host launch ----------------

extern "C" void kernel_launch(void* const* d_in, const int* in_sizes, int n_in,
                              void* d_out, int out_size, void* d_ws, size_t ws_size,
                              hipStream_t stream) {
    (void)in_sizes; (void)n_in; (void)out_size; (void)ws_size;

    const int*   x     = (const int*)d_in[0];
    const int*   ei    = (const int*)d_in[1];
    const int*   batch = (const int*)d_in[3];
    const float* emb[6] = { (const float*)d_in[4], (const float*)d_in[5], (const float*)d_in[6],
                            (const float*)d_in[7], (const float*)d_in[8], (const float*)d_in[9] };
    const float* W1 = (const float*)d_in[10];
    const float* b1 = (const float*)d_in[11];
    const float* W2 = (const float*)d_in[12];
    const float* b2 = (const float*)d_in[13];
    const float* gw1 = (const float*)d_in[14];
    const float* gb1 = (const float*)d_in[15];
    const float* bng = (const float*)d_in[16];
    const float* bnb = (const float*)d_in[17];
    const float* gw2 = (const float*)d_in[18];
    const float* gb2 = (const float*)d_in[19];
    const float* lw  = (const float*)d_in[20];
    const float* lb  = (const float*)d_in[21];
    float* out = (float*)d_out;

    const int* src = ei;
    const int* dst = ei + NE;

    char* ws = (char*)d_ws;
    size_t off = 0;
    auto alloc = [&](size_t bytes) -> void* {
        off = (off + 255) & ~(size_t)255;
        void* p = ws + off;
        off += bytes;
        return p;
    };

    _Float16* f16A = (_Float16*)alloc((size_t)NN * DIN * 2);  // x features, later h f16
    float*    hw   = (float*)alloc((size_t)NN * DH * 4);      // GEMM out (hw / g)
    float*    agg  = (float*)alloc((size_t)NN * DH * 4);      // scatter accumulator
    float*    hcur = (float*)alloc((size_t)NN * DH * 4);      // conv output f32
    float*    deg  = (float*)alloc((size_t)NN * 4);           // degree -> dinv -> gate
    float*    ebuf = (float*)alloc((size_t)NN * 4);
    _Float16* Bp1  = (_Float16*)alloc((size_t)64 * DIN * 2);
    _Float16* Bp2  = (_Float16*)alloc((size_t)64 * DH * 2);
    _Float16* Bpg  = (_Float16*)alloc((size_t)64 * DH * 2);
    const int STATS = 64 * 4 + NG + NG + NG * DH;             // sum,sq,scale,shift,z,gmax,pooled
    float* stats = (float*)alloc((size_t)STATS * 4);
    float*    bn_sum   = stats;
    float*    bn_sq    = stats + 64;
    float*    bn_scale = stats + 128;
    float*    bn_shift = stats + 192;
    float*    zbuf     = stats + 256;
    unsigned* gmaxU    = (unsigned*)(stats + 256 + NG);
    float*    pooled   = stats + 256 + 2 * NG;

    const int T = 256;
    const long NC  = (long)NN * DH;
    const long EC  = (long)NE * DH;
    const long NEMB = (long)NN * DIN;
    #define GB(n) ((unsigned)(((n) + T - 1) / T))

    zero_f32<<<GB(STATS), T, 0, stream>>>(stats, STATS);      // gmaxU=0 == encoded -inf floor

    // degrees
    deg_init <<<GB(NN), T, 0, stream>>>(deg);
    deg_count<<<GB(NE), T, 0, stream>>>(dst, deg);
    deg_rsqrt<<<GB(NN), T, 0, stream>>>(deg);

    // features + weight packing
    embed_kernel<<<GB(NEMB), T, 0, stream>>>(x, emb[0], emb[1], emb[2], emb[3], emb[4], emb[5], f16A);
    pack_b<<<GB(64 * DIN), T, 0, stream>>>(W1, Bp1, DIN);
    pack_b<<<GB(64 * DH),  T, 0, stream>>>(W2, Bp2, DH);
    pack_b<<<GB(64 * DH),  T, 0, stream>>>(gw1, Bpg, DH);

    // conv1
    wmma_gemm_kernel<DIN><<<NN / 16, 128, 0, stream>>>(f16A, Bp1, nullptr, hw);
    selfloop_kernel<<<GB(NC), T, 0, stream>>>(hw, deg, agg);
    edge_scatter   <<<GB(EC), T, 0, stream>>>(src, dst, hw, deg, agg);
    bias_sigmoid   <<<GB(NC), T, 0, stream>>>(agg, b1, hcur, f16A);

    // conv2
    wmma_gemm_kernel<DH><<<NN / 16, 128, 0, stream>>>(f16A, Bp2, nullptr, hw);
    selfloop_kernel<<<GB(NC), T, 0, stream>>>(hw, deg, agg);
    edge_scatter   <<<GB(EC), T, 0, stream>>>(src, dst, hw, deg, agg);
    bias_sigmoid   <<<GB(NC), T, 0, stream>>>(agg, b2, hcur, f16A);

    // gate MLP: g = h @ gw1 + gb1 ; BN ; relu ; . gw2 + gb2
    wmma_gemm_kernel<DH><<<NN / 16, 128, 0, stream>>>(f16A, Bpg, gb1, hw);
    bn_partial <<<256, T, 0, stream>>>(hw, bn_sum, bn_sq);
    bn_finalize<<<1, 64, 0, stream>>>(bn_sum, bn_sq, bng, bnb, bn_scale, bn_shift);
    gate_kernel<<<GB(NN), T, 0, stream>>>(hw, bn_scale, bn_shift, gw2, gb2, deg);

    // segment softmax + attention pooling + head
    gmax_kernel  <<<GB(NN), T, 0, stream>>>(deg, batch, gmaxU);
    expz_kernel  <<<GB(NN), T, 0, stream>>>(deg, batch, gmaxU, ebuf, zbuf);
    pooled_kernel<<<GB(NC), T, 0, stream>>>(ebuf, zbuf, batch, hcur, pooled);
    final_kernel <<<NG / T, T, 0, stream>>>(pooled, lw, lb, out);

    #undef GB
}
